// EncoderDecoder_55284819034882
// MI455X (gfx1250) — matrix operand order repaired and checked by reading, not compiled, not run on hardware
//
#include <hip/hip_runtime.h>
#include <hip/hip_bf16.h>

typedef _Float16 f16_t;
typedef __attribute__((ext_vector_type(16))) _Float16 v16h;
typedef __attribute__((ext_vector_type(8)))  float    v8f;

struct alignas(16) h8 { f16_t v[8]; };   // 16-byte vector copy unit

#define BB 16
#define SS 256
#define EE 768
#define HH 512
#define TT 100
#define VV 30522
#define H3 1536

// ---------------- fp32 -> fp16 convert (with column slice for out_W split) ----
__global__ void ed_cvt_f16(const float* __restrict__ src, long src_ld, long col0,
                           f16_t* __restrict__ dst, long total, long cols)
{
    long i = (long)blockIdx.x * blockDim.x + threadIdx.x;
    if (i >= total) return;
    long r = i / cols, c = i - r * cols;
    dst[i] = (f16_t)src[r * src_ld + col0 + c];
}

// ---------------- WMMA GEMM: out[m,n] = sum_k A[m,k] * W[n,k] (+bias +addm) ---
// A: [M x K] f16 row-major (M multiple of 16), W: [wRows x K] f16 row-major.
// Block = 128 threads = 4 waves computing a 16 x 128 output tile.
// The 16-row A panel is staged through LDS in 16x128-half chunks (read once per
// block from global, re-read via ds_load); each wave register-blocks 2 N-tiles,
// reusing one A fragment across 2 WMMAs. K must be a multiple of 128.
__global__ __launch_bounds__(128)
void ed_wmma_gemm(const f16_t* __restrict__ A, int lda,
                  const f16_t* __restrict__ W, int ldw, int wRows,
                  float* __restrict__ out, long ldo,
                  const float* __restrict__ bias,
                  const float* __restrict__ addm, int ldadd,
                  int N, int K)
{
    __shared__ alignas(16) f16_t tileA[16 * 128];

    const int lane = threadIdx.x & 31;
    const int wave = threadIdx.x >> 5;
    const int lo   = lane & 15;
    const int hi   = lane >> 4;
    const int mT   = blockIdx.y;

    // two 16-wide N tiles per wave
    const int nT0 = blockIdx.x * 8 + wave * 2;
    const int nT1 = nT0 + 1;
    int wr0 = nT0 * 16 + lo; if (wr0 >= wRows) wr0 = wRows - 1;
    int wr1 = nT1 * 16 + lo; if (wr1 >= wRows) wr1 = wRows - 1;
    const f16_t* wRow0 = W + (long)wr0 * ldw;
    const f16_t* wRow1 = W + (long)wr1 * ldw;

    // cooperative A-stage mapping: 16 rows x 8 segments of 16 halves
    const int sr  = threadIdx.x >> 3;         // 0..15  row
    const int sc  = (threadIdx.x & 7) * 16;   // 0..112 col (halves)
    const f16_t* aSrc = A + (long)(mT * 16 + sr) * lda + sc;

    v8f acc0 = {}, acc1 = {};
    for (int k0 = 0; k0 < K; k0 += 128) {
        __syncthreads();
        *(h8*)&tileA[sr * 128 + sc]     = *(const h8*)(aSrc + k0);
        *(h8*)&tileA[sr * 128 + sc + 8] = *(const h8*)(aSrc + k0 + 8);
        __syncthreads();
        // hint the next weight chunk toward the caches (global_prefetch_b8)
        if (k0 + 128 < K) {
            __builtin_prefetch(wRow0 + k0 + 128, 0, 1);
            __builtin_prefetch(wRow1 + k0 + 128, 0, 1);
        }
#pragma unroll
        for (int kk = 0; kk < 128; kk += 32) {
            // A fragment (contiguous k order, same as B): K = [kk+16*hi,+16)
            const f16_t* ap0 = &tileA[lo * 128 + kk + 16 * hi];
            const f16_t* ap1 = ap0 + 8;
            v16h a;
#pragma unroll
            for (int i = 0; i < 8; ++i) { a[i] = ap0[i]; a[i + 8] = ap1[i]; }
            // B fragments (contiguous k order, same as A): K = [kk+16*hi,+16) of column wr
            const f16_t* bp0 = wRow0 + k0 + kk + 16 * hi;
            const f16_t* bp1 = wRow1 + k0 + kk + 16 * hi;
            v16h b0, b1;
#pragma unroll
            for (int i = 0; i < 16; ++i) { b0[i] = bp0[i]; b1[i] = bp1[i]; }
            acc0 = __builtin_amdgcn_wmma_f32_16x16x32_f16(false, a, false, b0,
                                                          (short)0, acc0, false, false);
            acc1 = __builtin_amdgcn_wmma_f32_16x16x32_f16(false, a, false, b1,
                                                          (short)0, acc1, false, false);
        }
    }

    // store both tiles (C/D layout: M = r + 8*hi, N = lane&15)
    const int col0 = nT0 * 16 + lo;
    const int col1 = nT1 * 16 + lo;
#pragma unroll
    for (int r = 0; r < 8; ++r) {
        const int mrow = mT * 16 + r + 8 * hi;
        if (col0 < N) {
            float v = acc0[r];
            if (bias) v += bias[col0];
            if (addm) v += addm[(long)mrow * ldadd + col0];
            out[(long)mrow * ldo + col0] = v;
        }
        if (col1 < N) {
            float v = acc1[r];
            if (bias) v += bias[col1];
            if (addm) v += addm[(long)mrow * ldadd + col1];
            out[(long)mrow * ldo + col1] = v;
        }
    }
}

// ---------------- fused GRU gate/update ------------------------------------
__global__ void ed_gru_cell(const float* __restrict__ gi, long gi_stride,
                            const float* __restrict__ gh,
                            const float* __restrict__ bih, const float* __restrict__ bhh,
                            const float* __restrict__ hprev,
                            float* __restrict__ hnew, f16_t* __restrict__ hnew_h,
                            float* __restrict__ enc_store, long enc_stride)
{
    int idx = blockIdx.x * blockDim.x + threadIdx.x;
    if (idx >= BB * HH) return;
    int b = idx >> 9;           // / HH
    int j = idx & (HH - 1);
    const float* gib = gi + (long)b * gi_stride;
    const float* ghb = gh + (long)b * H3;
    float rr  = gib[j]          + bih[j]          + ghb[j]          + bhh[j];
    float zz  = gib[HH + j]     + bih[HH + j]     + ghb[HH + j]     + bhh[HH + j];
    float in_ = gib[2 * HH + j] + bih[2 * HH + j];
    float hn  = ghb[2 * HH + j] + bhh[2 * HH + j];
    float r = 1.f / (1.f + expf(-rr));
    float z = 1.f / (1.f + expf(-zz));
    float n = tanhf(in_ + r * hn);
    float h2 = (1.f - z) * n + z * hprev[idx];
    hnew[idx] = h2;
    hnew_h[idx] = (f16_t)h2;
    if (enc_store) enc_store[(long)b * enc_stride + j] = h2;
}

__global__ void ed_zero_h(float* a, f16_t* ah)
{
    int i = blockIdx.x * blockDim.x + threadIdx.x;
    if (i >= BB * HH) return;
    a[i] = 0.f; ah[i] = (f16_t)0.f;
}

__global__ void ed_hinit(const float* __restrict__ hf, const float* __restrict__ hb,
                         float* __restrict__ h, f16_t* __restrict__ hh)
{
    int i = blockIdx.x * blockDim.x + threadIdx.x;
    if (i >= BB * HH) return;
    float v = hf[i] + hb[i];
    h[i] = v; hh[i] = (f16_t)v;
}

// u[j] = sum_i attn_W[i, j] * comb_W[i]   (j < 2H; attention collapses to enc_out . u)
__global__ void ed_compute_u(const float* __restrict__ attnW, const float* __restrict__ combW,
                             float* __restrict__ u)
{
    int j = blockIdx.x * blockDim.x + threadIdx.x;
    if (j >= 2 * HH) return;
    float acc = 0.f;
    for (int i = 0; i < HH; ++i) acc += attnW[(long)i * H3 + j] * combW[i];
    u[j] = acc;
}

__global__ void ed_scores(const float* __restrict__ enc_out, const float* __restrict__ u,
                          const int* __restrict__ mask, float* __restrict__ scores)
{
    int idx = blockIdx.x * blockDim.x + threadIdx.x;
    if (idx >= BB * SS) return;
    if (mask[idx] == 0) { scores[idx] = -1e9f; return; }
    const float* row = enc_out + (long)idx * (2 * HH);
    float acc = 0.f;
    for (int d = 0; d < 2 * HH; ++d) acc += row[d] * u[d];
    scores[idx] = acc;
}

__global__ void ed_softmax(float* __restrict__ scores)   // one block per batch row, S threads
{
    __shared__ float red[SS];
    int b = blockIdx.x, s = threadIdx.x;
    float v = scores[b * SS + s];
    red[s] = v; __syncthreads();
    for (int off = SS / 2; off > 0; off >>= 1) {
        if (s < off) red[s] = fmaxf(red[s], red[s + off]);
        __syncthreads();
    }
    float mx = red[0]; __syncthreads();
    float e = expf(v - mx);
    red[s] = e; __syncthreads();
    for (int off = SS / 2; off > 0; off >>= 1) {
        if (s < off) red[s] += red[s + off];
        __syncthreads();
    }
    scores[b * SS + s] = e / red[0];
}

__global__ void ed_ctx(const float* __restrict__ w, const float* __restrict__ enc_out,
                       float* __restrict__ ctx, f16_t* __restrict__ ctx_h)
{
    int b = blockIdx.x;
    for (int d = threadIdx.x; d < 2 * HH; d += blockDim.x) {
        float acc = 0.f;
        for (int s = 0; s < SS; ++s)
            acc += w[b * SS + s] * enc_out[((long)(b * SS + s)) * (2 * HH) + d];
        ctx[b * 2 * HH + d] = acc;
        ctx_h[b * 2 * HH + d] = (f16_t)acc;
    }
}

// apply valid-length zero mask to projected ctx, emit f16 copy
__global__ void ed_finalize_ctx(float* __restrict__ ctxp, f16_t* __restrict__ ctxp_h,
                                const int* __restrict__ mask)
{
    __shared__ int red[SS];
    int b = blockIdx.x, t = threadIdx.x;
    red[t] = mask[b * SS + t]; __syncthreads();
    for (int off = SS / 2; off > 0; off >>= 1) {
        if (t < off) red[t] += red[t + off];
        __syncthreads();
    }
    int valid = red[0];
    for (int j = t; j < HH; j += blockDim.x) {
        float v = (valid > 0) ? ctxp[b * HH + j] : 0.f;
        ctxp[b * HH + j] = v;
        ctxp_h[b * HH + j] = (f16_t)v;
    }
}

// rnn_in[t*16+b, :] = [emb[ids[b,t]] (E), ctx_proj[b] (H)]  (teacher forcing)
__global__ void ed_build_rnn_in(const int* __restrict__ ids, const float* __restrict__ emb,
                                const f16_t* __restrict__ ctxp_h, f16_t* __restrict__ rnn_in)
{
    const long total = (long)(TT - 1) * BB * (EE + HH);
    long i = (long)blockIdx.x * blockDim.x + threadIdx.x;
    if (i >= total) return;
    long k = i % (EE + HH);
    long row = i / (EE + HH);
    int t = (int)(row / BB), b = (int)(row % BB);
    if (k < EE) {
        int tok = ids[b * TT + t];
        rnn_in[i] = (f16_t)emb[(long)tok * EE + k];
    } else {
        rnn_in[i] = ctxp_h[b * HH + (k - EE)];
    }
}

extern "C" void kernel_launch(void* const* d_in, const int* in_sizes, int n_in,
                              void* d_out, int out_size, void* d_ws, size_t ws_size,
                              hipStream_t stream)
{
    const float* bert  = (const float*)d_in[0];
    const int*   amask = (const int*)  d_in[1];
    const int*   ids   = (const int*)  d_in[2];
    const float* emb   = (const float*)d_in[3];
    const float* WihF  = (const float*)d_in[4];
    const float* WhhF  = (const float*)d_in[5];
    const float* bihF  = (const float*)d_in[6];
    const float* bhhF  = (const float*)d_in[7];
    const float* WihB  = (const float*)d_in[8];
    const float* WhhB  = (const float*)d_in[9];
    const float* bihB  = (const float*)d_in[10];
    const float* bhhB  = (const float*)d_in[11];
    const float* dWih  = (const float*)d_in[12];
    const float* dWhh  = (const float*)d_in[13];
    const float* dbih  = (const float*)d_in[14];
    const float* dbhh  = (const float*)d_in[15];
    const float* attnW = (const float*)d_in[16];
    // d_in[17] attn_b and d_in[19] comb_b cancel under softmax shift-invariance
    const float* combW = (const float*)d_in[18];
    const float* projW = (const float*)d_in[20];
    const float* projb = (const float*)d_in[21];
    const float* outW  = (const float*)d_in[22];
    const float* outb  = (const float*)d_in[23];
    float* out = (float*)d_out;

    char* base = (char*)d_ws;
    size_t off = 0;
    auto alloc = [&](size_t bytes) -> void* {
        void* p = base + off;
        off = (off + bytes + 255) & ~(size_t)255;
        return p;
    };

    // ---- f16 operand buffers ----
    f16_t* bert_h  = (f16_t*)alloc((size_t)BB * SS * EE * 2);
    f16_t* WihF_h  = (f16_t*)alloc((size_t)H3 * EE * 2);
    f16_t* WhhF_h  = (f16_t*)alloc((size_t)H3 * HH * 2);
    f16_t* WihB_h  = (f16_t*)alloc((size_t)H3 * EE * 2);
    f16_t* WhhB_h  = (f16_t*)alloc((size_t)H3 * HH * 2);
    f16_t* dWih_h  = (f16_t*)alloc((size_t)H3 * (EE + HH) * 2);
    f16_t* dWhh_h  = (f16_t*)alloc((size_t)H3 * HH * 2);
    f16_t* projW_h = (f16_t*)alloc((size_t)HH * 2 * HH * 2);
    f16_t* outW1_h = (f16_t*)alloc((size_t)VV * HH * 2);   // out_W[:, :H]
    f16_t* outW2_h = (f16_t*)alloc((size_t)VV * HH * 2);   // out_W[:, H:]
    f16_t* hF_h    = (f16_t*)alloc((size_t)BB * HH * 2);
    f16_t* hB_h    = (f16_t*)alloc((size_t)BB * HH * 2);
    f16_t* hD_h    = (f16_t*)alloc((size_t)BB * HH * 2);
    f16_t* ctx_h   = (f16_t*)alloc((size_t)BB * 2 * HH * 2);
    f16_t* ctxp_h  = (f16_t*)alloc((size_t)BB * HH * 2);
    f16_t* rnn_h   = (f16_t*)alloc((size_t)(TT - 1) * BB * (EE + HH) * 2);
    // ---- f32 buffers ----
    float* giF     = (float*)alloc((size_t)BB * SS * H3 * 4);
    float* giB     = (float*)alloc((size_t)BB * SS * H3 * 4);
    float* giD     = (float*)alloc((size_t)(TT - 1) * BB * H3 * 4);
    float* ghF     = (float*)alloc((size_t)BB * H3 * 4);
    float* ghB     = (float*)alloc((size_t)BB * H3 * 4);
    float* ghD     = (float*)alloc((size_t)BB * H3 * 4);
    float* enc_out = (float*)alloc((size_t)BB * SS * 2 * HH * 4);
    float* hF      = (float*)alloc((size_t)BB * HH * 4);
    float* hB      = (float*)alloc((size_t)BB * HH * 4);
    float* hD      = (float*)alloc((size_t)BB * HH * 4);
    float* uvec    = (float*)alloc((size_t)2 * HH * 4);
    float* scores  = (float*)alloc((size_t)BB * SS * 4);
    float* ctx     = (float*)alloc((size_t)BB * 2 * HH * 4);
    float* ctxp    = (float*)alloc((size_t)BB * HH * 4);
    float* lbase   = (float*)alloc((size_t)BB * VV * 4);
    (void)ws_size; (void)in_sizes; (void)n_in; (void)out_size;

    auto cvt = [&](const float* s, long ld, long c0, f16_t* d, long rows, long cols) {
        long tot = rows * cols;
        ed_cvt_f16<<<dim3((unsigned)((tot + 255) / 256)), dim3(256), 0, stream>>>(s, ld, c0, d, tot, cols);
    };
    // weight/activation conversions
    cvt(bert, EE, 0, bert_h, (long)BB * SS, EE);
    cvt(WihF, EE, 0, WihF_h, H3, EE);
    cvt(WhhF, HH, 0, WhhF_h, H3, HH);
    cvt(WihB, EE, 0, WihB_h, H3, EE);
    cvt(WhhB, HH, 0, WhhB_h, H3, HH);
    cvt(dWih, EE + HH, 0, dWih_h, H3, EE + HH);
    cvt(dWhh, HH, 0, dWhh_h, H3, HH);
    cvt(projW, 2 * HH, 0, projW_h, HH, 2 * HH);
    cvt(outW, 2 * HH, 0,  outW1_h, VV, HH);
    cvt(outW, 2 * HH, HH, outW2_h, VV, HH);

    ed_zero_h<<<dim3(32), dim3(256), 0, stream>>>(hF, hF_h);
    ed_zero_h<<<dim3(32), dim3(256), 0, stream>>>(hB, hB_h);

    const unsigned nb1536 = (H3 + 127) / 128;        // 12
    const unsigned nb512  = (HH + 127) / 128;        // 4
    const unsigned nbV    = (VV + 127) / 128;        // 239

    // hoisted input-gate GEMMs: gi = x @ Wih^T  for all (b,s)
    ed_wmma_gemm<<<dim3(nb1536, (BB * SS) / 16), dim3(128), 0, stream>>>(
        bert_h, EE, WihF_h, EE, H3, giF, H3, nullptr, nullptr, 0, H3, EE);
    ed_wmma_gemm<<<dim3(nb1536, (BB * SS) / 16), dim3(128), 0, stream>>>(
        bert_h, EE, WihB_h, EE, H3, giB, H3, nullptr, nullptr, 0, H3, EE);

    // bidirectional recurrent scan: per step only h @ Whh^T + fused gates
    for (int s = 0; s < SS; ++s) {
        ed_wmma_gemm<<<dim3(nb1536, 1), dim3(128), 0, stream>>>(
            hF_h, HH, WhhF_h, HH, H3, ghF, H3, nullptr, nullptr, 0, H3, HH);
        ed_gru_cell<<<dim3(32), dim3(256), 0, stream>>>(
            giF + (long)s * H3, (long)SS * H3, ghF, bihF, bhhF,
            hF, hF, hF_h, enc_out + (long)s * 2 * HH, (long)SS * 2 * HH);
        int sb = SS - 1 - s;
        ed_wmma_gemm<<<dim3(nb1536, 1), dim3(128), 0, stream>>>(
            hB_h, HH, WhhB_h, HH, H3, ghB, H3, nullptr, nullptr, 0, H3, HH);
        ed_gru_cell<<<dim3(32), dim3(256), 0, stream>>>(
            giB + (long)sb * H3, (long)SS * H3, ghB, bihB, bhhB,
            hB, hB, hB_h, enc_out + (long)sb * 2 * HH + HH, (long)SS * 2 * HH);
    }
    ed_hinit<<<dim3(32), dim3(256), 0, stream>>>(hF, hB, hD, hD_h);

    // attention: collapses to a single masked softmax over enc_out . u (step-invariant)
    ed_compute_u<<<dim3(4), dim3(256), 0, stream>>>(attnW, combW, uvec);
    ed_scores<<<dim3(16), dim3(256), 0, stream>>>(enc_out, uvec, amask, scores);
    ed_softmax<<<dim3(BB), dim3(SS), 0, stream>>>(scores);
    ed_ctx<<<dim3(BB), dim3(256), 0, stream>>>(scores, enc_out, ctx, ctx_h);
    ed_wmma_gemm<<<dim3(nb512, 1), dim3(128), 0, stream>>>(          // ctx_proj
        ctx_h, 2 * HH, projW_h, 2 * HH, HH, ctxp, HH, projb, nullptr, 0, HH, 2 * HH);
    ed_finalize_ctx<<<dim3(BB), dim3(SS), 0, stream>>>(ctxp, ctxp_h, amask);

    // logits base: ctx_proj @ out_W[:,H:]^T + out_b (once for all 99 steps)
    ed_wmma_gemm<<<dim3(nbV, 1), dim3(128), 0, stream>>>(
        ctxp_h, HH, outW2_h, HH, VV, lbase, VV, outb, nullptr, 0, VV, HH);

    // hoisted decoder input-gate GEMM over all teacher-forced steps
    {
        long tot = (long)(TT - 1) * BB * (EE + HH);
        ed_build_rnn_in<<<dim3((unsigned)((tot + 255) / 256)), dim3(256), 0, stream>>>(
            ids, emb, ctxp_h, rnn_h);
    }
    ed_wmma_gemm<<<dim3(nb1536, ((TT - 1) * BB) / 16), dim3(128), 0, stream>>>(
        rnn_h, EE + HH, dWih_h, EE + HH, H3, giD, H3, nullptr, nullptr, 0, H3, EE + HH);

    // decoder scan: recurrent GEMM + gates + L2-resident vocab projection
    for (int t = 0; t < TT - 1; ++t) {
        ed_wmma_gemm<<<dim3(nb1536, 1), dim3(128), 0, stream>>>(
            hD_h, HH, dWhh_h, HH, H3, ghD, H3, nullptr, nullptr, 0, H3, HH);
        ed_gru_cell<<<dim3(32), dim3(256), 0, stream>>>(
            giD + (long)t * BB * H3, (long)H3, ghD, dbih, dbhh,
            hD, hD, hD_h, nullptr, 0);
        ed_wmma_gemm<<<dim3(nbV, 1), dim3(128), 0, stream>>>(
            hD_h, HH, outW1_h, HH, VV,
            out + (long)t * VV, (long)(TT - 1) * VV,
            nullptr, lbase, VV, VV, HH);
    }
}
